// FAGCN_68143951118602
// MI455X (gfx1250) — matrix operand order repaired and checked by reading, not compiled, not run on hardware
//
#include <hip/hip_runtime.h>
#include <hip/hip_bf16.h>

typedef __attribute__((ext_vector_type(16))) _Float16 v16h;
typedef __attribute__((ext_vector_type(8)))  float    v8f;
typedef __attribute__((ext_vector_type(4)))  unsigned int v4u;
typedef __attribute__((ext_vector_type(8)))  int      v8i;
typedef __attribute__((ext_vector_type(4)))  int      v4i;

#define HID 128
#define INDIM 256
#define OUTD 64

// ---------------------------------------------------------------------------
// TDM: async 2D tile load Global -> LDS (CDNA5 ISA).
// D# group0: [1:0]count=1, [63:32]lds_addr, [120:64]global_addr, [127:126]type=2
// D# group1: [17:16]data_size(2=4B), [79:48]tensor_dim0, [111:80]tensor_dim1,
//            [127:112]tile_dim0, [143:128]tile_dim1, [207:160]tensor_dim0_stride
// Groups 2/3 (and the trailing int32x8 operand) zero for a 2D tensor.
// Tracked by TENSORcnt. This toolchain exposes the 6-arg builtin form.
// ---------------------------------------------------------------------------
#if __has_builtin(__builtin_amdgcn_tensor_load_to_lds)
#define HAVE_TDM 1
__device__ __forceinline__ void tdm_load_2d(const void* gaddr, unsigned lds_off,
                                            unsigned tensor_d0, unsigned tensor_d1,
                                            unsigned tile_d0, unsigned tile_d1,
                                            unsigned stride0_elems) {
    const unsigned long long ga = (unsigned long long)(uintptr_t)gaddr;
    v4u g0;
    g0[0] = 1u;                                        // count=1 (user descriptor)
    g0[1] = lds_off;                                   // LDS byte address
    g0[2] = (unsigned)(ga & 0xFFFFFFFFu);
    g0[3] = (unsigned)((ga >> 32) & 0x1FFFFFFu) | (2u << 30);  // type=2 "image"
    v8i g1;
    g1[0] = (int)(2u << 16);                           // data_size = 4 bytes
    g1[1] = (int)((tensor_d0 & 0xFFFFu) << 16);        // tensor_dim0[15:0]
    g1[2] = (int)((tensor_d0 >> 16) | ((tensor_d1 & 0xFFFFu) << 16));
    g1[3] = (int)((tensor_d1 >> 16) | ((tile_d0 & 0xFFFFu) << 16));
    g1[4] = (int)(tile_d1 & 0xFFFFu);                  // tile_dim1 (tile_dim2=0)
    g1[5] = (int)stride0_elems;                        // tensor_dim0_stride[31:0]
    g1[6] = 0;                                         // stride0[47:32]=0, stride1=0
    g1[7] = 0;
    v4i z4 = {0, 0, 0, 0};
    v8i z8 = {0, 0, 0, 0, 0, 0, 0, 0};
    __builtin_amdgcn_tensor_load_to_lds(g0, g1, z4, z4, z8, 0);
}
#else
#define HAVE_TDM 0
#endif

// Drain outstanding LDS reads before the async engine may overwrite the slice.
__device__ __forceinline__ void wait_ds_zero() {
#if __has_builtin(__builtin_amdgcn_s_wait_dscnt)
    __builtin_amdgcn_s_wait_dscnt(0);
#else
    asm volatile("s_wait_dscnt 0" ::: "memory");
#endif
}

// ---------------------------------------------------------------------------
// Weight pre-pass: W[K][NC] f32 -> Wt[NC][K] f16 (transposed, WMMA-B friendly:
// each lane's 16-half B fragment becomes two contiguous 16-byte loads).
// ---------------------------------------------------------------------------
__global__ void convert_weights(const float* __restrict__ W,
                                _Float16* __restrict__ Wt, int K, int NC) {
    const int i = blockIdx.x * blockDim.x + threadIdx.x;
    if (i >= K * NC) return;
    const int k = i / NC, n = i - k * NC;
    Wt[(size_t)n * K + k] = (_Float16)W[i];
}

// ---------------------------------------------------------------------------
// GEMM: C[M,NC] = act(A[M,K] @ B[K,NC] + bias).
// Block = 256 threads = 8 waves; wave w owns the 16x16 tile at
// (blockIdx.x*8 + w, blockIdx.y). Double-buffered TDM pipeline per K-step:
//   - TDM prefetches the NEXT 16x32 f32 A-tile into the other 2KB LDS slice
//   - s_wait_tensorcnt(1) => current tile landed (TENSORcnt is in-order)
//   - compiler memory barrier, then ds_load_b128 fragments, cvt -> f16
//   - B fragment = two 16-byte loads (K+0..7, K+16..23) of pre-transposed f16 weights
//   - v_wmma_f32_16x16x32_f16 accumulate (layouts per ISA 7.12.2)
// ---------------------------------------------------------------------------
__global__ void gemm_wmma_bias_act(const float* __restrict__ A,
                                   const _Float16* __restrict__ Bt,  // [NC][K]
                                   const float* __restrict__ bias,
                                   float* __restrict__ C,
                                   int M, int K, int NC, int do_relu) {
#if HAVE_TDM
    __shared__ float Atile[8][2][16][32];              // 32 KB, ping-pong per wave
#endif
    const int wid  = threadIdx.x >> 5;
    const int lane = threadIdx.x & 31;
    const int tm   = blockIdx.x * 8 + wid;             // 16-row tile index
    if (tm * 16 >= M) return;                          // wave-uniform exit
    const int tn   = blockIdx.y;

    const int m16  = lane & 15;
    const int ncol = tn * 16 + m16;
    const int kbA  = (lane < 16) ? 0 : 8;              // A K sub-base
    const int kbB  = (lane < 16) ? 0 : 8;              // B K sub-base

    const float*    Abase = A  + (size_t)tm * 16 * K;
    const _Float16* Bbase = Bt + (size_t)ncol * K;

#if HAVE_TDM
    const unsigned lds0 = (unsigned)(uintptr_t)(&Atile[wid][0][0][0]);
    const unsigned lds1 = (unsigned)(uintptr_t)(&Atile[wid][1][0][0]);
    // Prologue: DMA first tile into slice 0.
    tdm_load_2d(Abase, lds0, (unsigned)K, 16u, 32u, 16u, (unsigned)K);
#endif

    v8f acc = {};
    for (int k0 = 0; k0 < K; k0 += 32) {
        // B fragment: elems 0-7 -> K = k0+kbB..+7, elems 8-15 -> K = k0+kbB+16..+23.
        typedef __attribute__((ext_vector_type(8))) _Float16 v8h_;
        const v8h_ b0 = *(const v8h_*)(Bbase + k0 + kbB);
        const v8h_ b1 = *(const v8h_*)(Bbase + k0 + kbB + 16);
        const v16h b = __builtin_shufflevector(b0, b1, 0,1,2,3,4,5,6,7,8,9,10,11,12,13,14,15);

#if HAVE_TDM
        const int cur = (k0 >> 5) & 1;
        if (k0 + 32 < K) {
            // WAR guard: prior ds_loads from the target slice must be complete.
            wait_ds_zero();
            tdm_load_2d(Abase + (k0 + 32), cur ? lds0 : lds1,
                        (unsigned)K, 16u, 32u, 16u, (unsigned)K);
            __builtin_amdgcn_s_wait_tensorcnt(1);      // current tile landed
        } else {
            __builtin_amdgcn_s_wait_tensorcnt(0);      // last tile landed
        }
        // TDM writes LDS behind the compiler's back: force re-load of the tile.
        asm volatile("" ::: "memory");
        const float* arow = &Atile[wid][cur][m16][0];
#else
        const float* arow = Abase + (size_t)m16 * K + k0;
#endif
        // A fragment: elems 0-7 -> K=kbA..kbA+7, elems 8-15 -> K=kbA+16..kbA+23
        const float4 x0 = *(const float4*)(arow + kbA);
        const float4 x1 = *(const float4*)(arow + kbA + 4);
        const float4 x2 = *(const float4*)(arow + kbA + 16);
        const float4 x3 = *(const float4*)(arow + kbA + 20);
        v16h a;
        a[0]  = (_Float16)x0.x; a[1]  = (_Float16)x0.y;
        a[2]  = (_Float16)x0.z; a[3]  = (_Float16)x0.w;
        a[4]  = (_Float16)x1.x; a[5]  = (_Float16)x1.y;
        a[6]  = (_Float16)x1.z; a[7]  = (_Float16)x1.w;
        a[8]  = (_Float16)x2.x; a[9]  = (_Float16)x2.y;
        a[10] = (_Float16)x2.z; a[11] = (_Float16)x2.w;
        a[12] = (_Float16)x3.x; a[13] = (_Float16)x3.y;
        a[14] = (_Float16)x3.z; a[15] = (_Float16)x3.w;

        acc = __builtin_amdgcn_wmma_f32_16x16x32_f16(
            false, a, false, b, (short)0, acc, false, false);
    }

    // C/D layout: element r -> M = tm*16 + (lane<16 ? r : 8+r), N = ncol.
    const int mbase = tm * 16 + ((lane < 16) ? 0 : 8);
    const float bv = bias[ncol];
#pragma unroll
    for (int r = 0; r < 8; ++r) {
        float v = acc[r] + bv;
        if (do_relu) v = fmaxf(v, 0.0f);
        C[(size_t)(mbase + r) * NC + ncol] = v;
    }
}

// ---------------------------------------------------------------------------
// Per-node attention partials: s_src[n]=h[n]·w[0:128], s_dst[n]=h[n]·w[128:256].
// ---------------------------------------------------------------------------
__global__ void node_scores(const float* __restrict__ h,
                            const float* __restrict__ att_w,
                            float* __restrict__ s_src,
                            float* __restrict__ s_dst, int N) {
    const int node = blockIdx.x * (blockDim.x >> 5) + (threadIdx.x >> 5);
    const int lane = threadIdx.x & 31;
    if (node >= N) return;
    const float* hn = h + (size_t)node * HID;
    float a = 0.0f, b = 0.0f;
#pragma unroll
    for (int k = lane; k < HID; k += 32) {
        const float v = hn[k];
        a += v * att_w[k];
        b += v * att_w[HID + k];
    }
#pragma unroll
    for (int off = 16; off > 0; off >>= 1) {
        a += __shfl_down(a, off, 32);
        b += __shfl_down(b, off, 32);
    }
    if (lane == 0) { s_src[node] = a; s_dst[node] = b; }
}

// ---------------------------------------------------------------------------
// alpha[e] = tanh(s_src[row[e]] + s_dst[col[e]] + b)  (once per edge)
// ---------------------------------------------------------------------------
__global__ void edge_alpha(const int* __restrict__ row, const int* __restrict__ col,
                           const float* __restrict__ s_src,
                           const float* __restrict__ s_dst,
                           const float* __restrict__ att_b,
                           float* __restrict__ alpha, int E) {
    const int e = blockIdx.x * blockDim.x + threadIdx.x;
    if (e >= E) return;
    alpha[e] = tanhf(s_src[row[e]] + s_dst[col[e]] + att_b[0]);
}

// ---------------------------------------------------------------------------
// One wave per edge: 4 features/lane, float4 gather of h[row], 4 hardware
// f32 atomics into L2-resident agg[col].
// ---------------------------------------------------------------------------
__global__ void edge_propagate(const float* __restrict__ h,
                               const int* __restrict__ row,
                               const int* __restrict__ col,
                               const float* __restrict__ alpha,
                               float* __restrict__ agg, long long E) {
    const long long tid = (long long)blockIdx.x * blockDim.x + threadIdx.x;
    if (tid >= E * 32) return;
    const int e = (int)(tid >> 5);
    const int k = (int)(tid & 31) * 4;
    const int r = row[e];
    const int c = col[e];
    const float  al = alpha[e];
    const float4 hv = *(const float4*)(h + (size_t)r * HID + k);
    float* ap = agg + (size_t)c * HID + k;
    unsafeAtomicAdd(ap + 0, al * hv.x);
    unsafeAtomicAdd(ap + 1, al * hv.y);
    unsafeAtomicAdd(ap + 2, al * hv.z);
    unsafeAtomicAdd(ap + 3, al * hv.w);
}

// ---------------------------------------------------------------------------
// agg[i] = relu(eps*hin[i] + (1-eps)*agg[i])
// ---------------------------------------------------------------------------
__global__ void combine_relu(const float* __restrict__ hin,
                             float* __restrict__ agg,
                             const float* __restrict__ eps_p, long long total) {
    const long long i = (long long)blockIdx.x * blockDim.x + threadIdx.x;
    if (i >= total) return;
    const float eps = eps_p[0];
    agg[i] = fmaxf(eps * hin[i] + (1.0f - eps) * agg[i], 0.0f);
}

extern "C" void kernel_launch(void* const* d_in, const int* in_sizes, int n_in,
                              void* d_out, int out_size, void* d_ws, size_t ws_size,
                              hipStream_t stream) {
    const float* x      = (const float*)d_in[0];
    const int*   eidx   = (const int*)  d_in[1];   // [2,E] int32 (JAX x64 off)
    const float* W0     = (const float*)d_in[2];
    const float* b0     = (const float*)d_in[3];
    const float* att_w1 = (const float*)d_in[4];
    const float* att_b1 = (const float*)d_in[5];
    const float* eps1   = (const float*)d_in[6];
    const float* att_w2 = (const float*)d_in[7];
    const float* att_b2 = (const float*)d_in[8];
    const float* eps2   = (const float*)d_in[9];
    const float* Wc     = (const float*)d_in[10];
    const float* bc     = (const float*)d_in[11];
    float* out = (float*)d_out;

    const int N = in_sizes[0] / INDIM;          // 50000
    const long long E = in_sizes[1] / 2;        // 800000
    const int* rowp = eidx;
    const int* colp = eidx + E;

    // Workspace layout
    float*    hA     = (float*)d_ws;                        // N*HID
    float*    hB     = hA + (size_t)N * HID;                // N*HID
    float*    s_src  = hB + (size_t)N * HID;                // N
    float*    s_dst  = s_src + N;                           // N
    float*    alphaB = s_dst + N;                           // E
    _Float16* W0t    = (_Float16*)(alphaB + E);             // NC0*K0 = 128*256
    _Float16* Wct    = W0t + (size_t)HID * INDIM;           // 64*128

    const long long nh = (long long)N * HID;

    // Weight pre-transpose + f32->f16 (tiny, once per launch)
    convert_weights<<<(INDIM * HID + 255) / 256, 256, 0, stream>>>(W0, W0t, INDIM, HID);
    convert_weights<<<(HID * OUTD + 255) / 256, 256, 0, stream>>>(Wc, Wct, HID, OUTD);

    // 1) h0 = relu(x @ W0 + b0) -> hA
    {
        dim3 grid((N / 16 + 7) / 8, HID / 16);
        gemm_wmma_bias_act<<<grid, 256, 0, stream>>>(x, W0t, b0, hA, N, INDIM, HID, 1);
    }

    // ---- FAGCN layer 1 ----
    {
        node_scores<<<(N + 7) / 8, 256, 0, stream>>>(hA, att_w1, s_src, s_dst, N);
        edge_alpha<<<(int)((E + 255) / 256), 256, 0, stream>>>(
            rowp, colp, s_src, s_dst, att_b1, alphaB, (int)E);
        (void)hipMemsetAsync(hB, 0, (size_t)nh * sizeof(float), stream);
        const long long tot = E * 32;
        edge_propagate<<<(int)((tot + 255) / 256), 256, 0, stream>>>(
            hA, rowp, colp, alphaB, hB, E);
        combine_relu<<<(int)((nh + 255) / 256), 256, 0, stream>>>(hA, hB, eps1, nh);
        // h1 in hB
    }

    // ---- FAGCN layer 2 ----
    {
        node_scores<<<(N + 7) / 8, 256, 0, stream>>>(hB, att_w2, s_src, s_dst, N);
        edge_alpha<<<(int)((E + 255) / 256), 256, 0, stream>>>(
            rowp, colp, s_src, s_dst, att_b2, alphaB, (int)E);
        (void)hipMemsetAsync(hA, 0, (size_t)nh * sizeof(float), stream);
        const long long tot = E * 32;
        edge_propagate<<<(int)((tot + 255) / 256), 256, 0, stream>>>(
            hB, rowp, colp, alphaB, hA, E);
        combine_relu<<<(int)((nh + 255) / 256), 256, 0, stream>>>(hB, hA, eps2, nh);
        // h2 in hA
    }

    // 2) out = h2 @ Wc + bc
    {
        dim3 grid((N / 16 + 7) / 8, OUTD / 16);
        gemm_wmma_bias_act<<<grid, 256, 0, stream>>>(hA, Wct, bc, out, N, HID, OUTD, 0);
    }
}
